// XConv_71313636982811
// MI455X (gfx1250) — matrix operand order repaired and checked by reading, not compiled, not run on hardware
//
#include <hip/hip_runtime.h>
#include <math.h>

// ---------------------------------------------------------------------------
// CDNA5 / gfx1250 XConv implementation, wave32 + v_wmma_f32_16x16x32_bf16.
// B operands pre-swizzled into fragment order (1 contiguous 32B load/lane);
// A operands fetched as two ds_load_b128 per lane from bf16 LDS tiles.
// Dual-accumulator WMMA scheduling to hide D->A/B RAW hazards.
// ---------------------------------------------------------------------------

typedef __bf16 bf16_t;
typedef bf16_t v16bf __attribute__((ext_vector_type(16)));
typedef float  v8f   __attribute__((ext_vector_type(8)));

__device__ __forceinline__ unsigned short f2bf(float f) {
  return __builtin_bit_cast(unsigned short, (__bf16)f);  // hw v_cvt_*bf16_f32
}

union BF16Frag { v16bf v; unsigned short u[16]; uint4 q[2]; };

// Generic (gather) A fragment: u[j<8] -> k = 8*half + j, u[j>=8] -> k = 16+8*half+(j-8)
template <typename F>
__device__ __forceinline__ v16bf make_a_frag(F f) {
  const int lane = threadIdx.x & 31;
  const int m = lane & 15, half = lane >> 4;
  BF16Frag r;
#pragma unroll
  for (int j = 0; j < 8; ++j) {
    r.u[j]     = f(m, 8 * half + j);
    r.u[8 + j] = f(m, 16 + 8 * half + j);
  }
  return r.v;
}

// Generic (gather) B fragment: u[j<8] -> k = 8*half + j, u[j>=8] -> k = 16+8*half+(j-8), n = lane%16
template <typename F>
__device__ __forceinline__ v16bf make_b_frag(F f) {
  const int lane = threadIdx.x & 31;
  const int n = lane & 15, half = lane >> 4;
  BF16Frag r;
#pragma unroll
  for (int j = 0; j < 16; ++j) r.u[j] = f(8 * half + j + (j >= 8 ? 8 : 0), n);
  return r.v;
}

// Fast A fragment from row-major bf16 LDS tile: two 16B ds_load_b128 per lane.
__device__ __forceinline__ v16bf lds_a_frag(const unsigned short* lds,
                                            int stride, int koff) {
  const int lane = threadIdx.x & 31;
  const int m = lane & 15, half = lane >> 4;
  const unsigned short* base = lds + m * stride + koff + 8 * half;
  BF16Frag r;
  r.q[0] = *(const uint4*)(base);
  r.q[1] = *(const uint4*)(base + 16);
  return r.v;
}

// Fast A fragment for K=16 (upper 16 K zero): one ds_load_b128 per lane.
__device__ __forceinline__ v16bf lds_a_frag_k16(const unsigned short* lds,
                                               int stride) {
  const int lane = threadIdx.x & 31;
  const int m = lane & 15, half = lane >> 4;
  BF16Frag r;
  r.q[0] = *(const uint4*)(lds + m * stride + 8 * half);
  uint4 z; z.x = z.y = z.z = z.w = 0u;
  r.q[1] = z;
  return r.v;
}

// Fast B fragment: weights pre-swizzled so each lane's 16 bf16 are contiguous.
__device__ __forceinline__ v16bf load_b_tile(const unsigned short* base, int tile) {
  return *(const v16bf*)(base + (tile << 9) + ((threadIdx.x & 31) << 4));
}

#define WMMA_BF16(acc, a, b)                                                   \
  acc = __builtin_amdgcn_wmma_f32_16x16x32_bf16(false, (a), false, (b),        \
                                                (short)0, (acc), false, false)

// C-tile store with bias (+ optional relu) into bf16 LDS, row-scattered.
template <bool RELU>
__device__ __forceinline__ void store_c(unsigned short* dst, int stride, int c0,
                                        const v8f& acc, const float* __restrict__ bias) {
  const int lane = threadIdx.x & 31, ln = lane & 15, half = lane >> 4;
  const int c = c0 + ln;
  const float bb = bias[c];
#pragma unroll
  for (int v = 0; v < 8; ++v) {
    float val = acc[v] + bb;
    if (RELU) val = fmaxf(val, 0.0f);
    dst[(v + 8 * half) * stride + c] = f2bf(val);
  }
}

// ---------------------------------------------------------------------------
// Workspace layout. bf16 weights stored in WMMA-B fragment order:
// per (kc,nt) 32x16 tile: 32 lanes x 16 bf16 contiguous (1KB per tile).
//   Bfc1 [K32 xN64 ]  @ ushort 0
//   Bfc2 [K64 xN64 ]  @ 2048
//   Bxc  [K64 xN256]  @ 6144
//   Bx1  [K256xN256]  @ 22528
//   Bx2  [K256xN256]  @ 88064
//   Bpw  [K512xN128]  @ 153600  -> end 219136 ushorts = 438272 bytes
#define OFF_STAT_BYTES 438272      // 256 floats: scale[128], shift[128]
#define OFF_PART_BYTES 439296      // 1024 tiles * 128 ch * {sum,sumsq} f32
#define N_WCONV_ELEMS  219136

__device__ __forceinline__ void frag_decode(int e, int NT, int& k, int& n) {
  int tile = e >> 9, r = e & 511, lane = r >> 4, j = r & 15;
  int kc = tile / NT, nt = tile - kc * NT;
  k = kc * 32 + ((lane >> 4) << 3) + j + (j >= 8 ? 8 : 0);   // 8*half + j (+8 for j>=8)
  n = nt * 16 + (lane & 15);
}

__global__ void prep_weights(const float* __restrict__ Wfc1,
                             const float* __restrict__ Wfc2,
                             const float* __restrict__ Wxc,
                             const float* __restrict__ Wx1,
                             const float* __restrict__ Wx2,
                             const float* __restrict__ Wpw,
                             unsigned short* __restrict__ ws) {
  int i = blockIdx.x * 256 + threadIdx.x;
  if (i >= N_WCONV_ELEMS) return;
  int k, n;
  if (i < 2048) {                                   // Bfc1 (K pad 3->32)
    frag_decode(i, 4, k, n);
    ws[i] = f2bf(k < 3 ? Wfc1[n * 3 + k] : 0.0f);
    return;
  }
  i -= 2048;
  if (i < 4096) {                                   // Bfc2
    frag_decode(i, 4, k, n);
    ws[2048 + i] = f2bf(Wfc2[n * 64 + k]);
    return;
  }
  i -= 4096;
  if (i < 16384) {                                  // Bxc (K=k*3+d, pad 48->64)
    frag_decode(i, 16, k, n);
    float v = 0.0f;
    if (k < 48) { int kk = k / 3, d = k - 3 * kk; v = Wxc[n * 48 + d * 16 + kk]; }
    ws[6144 + i] = f2bf(v);
    return;
  }
  i -= 16384;
  if (i < 65536) {                                  // Bx1
    frag_decode(i, 16, k, n);
    ws[22528 + i] = f2bf(Wx1[n * 256 + k]);
    return;
  }
  i -= 65536;
  if (i < 65536) {                                  // Bx2
    frag_decode(i, 16, k, n);
    ws[88064 + i] = f2bf(Wx2[n * 256 + k]);
    return;
  }
  i -= 65536;
  {                                                 // Bpw
    frag_decode(i, 8, k, n);
    ws[153600 + i] = f2bf(Wpw[n * 512 + k]);
  }
}

// ---------------------------------------------------------------------------
// Main fused kernel: one block = 8 wave32 = one tile of 16 points.
// ---------------------------------------------------------------------------
__global__ __launch_bounds__(256, 2)
void xconv_main(const float* __restrict__ rep_pt, const float* __restrict__ pts,
                const float* __restrict__ fts,
                const float* __restrict__ b_fc1, const float* __restrict__ b_fc2,
                const float* __restrict__ b_xconv,
                const float* __restrict__ b_xfc1, const float* __restrict__ b_xfc2,
                const float* __restrict__ W_dw, const float* __restrict__ b_dw,
                const float* __restrict__ b_pw,
                const unsigned short* __restrict__ wsw,
                float* __restrict__ partials, float* __restrict__ out) {
  // LDS (58 KB total, hand-overlaid, all 16B aligned):
  __shared__ __align__(16) unsigned short s_ptsl[16 * 64];  // bf16 coords, pad 48->64  2 KB
  __shared__ __align__(16) unsigned short s_X0[16 * 256];   // X0 / final X bf16        8 KB
  __shared__ __align__(16) unsigned short s_RB[8 * 1024];   // X1 then per-wave h      16 KB
  __shared__ __align__(16) unsigned short s_lb[8 * 1024];   // lifted fts, per wave    16 KB
  __shared__ __align__(16) unsigned short s_dw[16 * 512];   // depthwise out bf16      16 KB

  const int tid  = threadIdx.x;
  const int w    = tid >> 5;
  const int lane = tid & 31;
  const int ln   = lane & 15;
  const int half = lane >> 4;
  const int tile = blockIdx.x;
  const int gp0  = tile << 4;

  const unsigned short* Bfc1 = wsw;
  const unsigned short* Bfc2 = wsw + 2048;
  const unsigned short* Bxc  = wsw + 6144;
  const unsigned short* Bx1  = wsw + 22528;
  const unsigned short* Bx2  = wsw + 88064;
  const unsigned short* Bpw  = wsw + 153600;

  // ---- Stage 0: local coordinates -> bf16, zero-padded --------------------
  for (int idx = tid; idx < 16 * 64; idx += 256) {
    int pt = idx >> 6, r = idx & 63;
    unsigned short v = 0;
    if (r < 48) {
      int k = r / 3, d = r - 3 * k;
      int gpt = gp0 + pt;
      v = f2bf(pts[(gpt * 16 + k) * 3 + d] - rep_pt[gpt * 3 + d]);
    }
    s_ptsl[idx] = v;
  }
  __syncthreads();

  // ---- Stage 3: X0 = relu(ptsl(16x48) @ Wxconv(48x256) + b) ---------------
  {
    v8f a0 = {}, a1 = {};                       // tiles w and w+8
    for (int kc = 0; kc < 2; ++kc) {
      v16bf a = lds_a_frag(s_ptsl, 64, kc * 32);
      WMMA_BF16(a0, a, load_b_tile(Bxc, kc * 16 + w));
      WMMA_BF16(a1, a, load_b_tile(Bxc, kc * 16 + w + 8));
    }
    store_c<true>(s_X0, 256, w * 16, a0, b_xconv);
    store_c<true>(s_X0, 256, (w + 8) * 16, a1, b_xconv);
  }
  __syncthreads();

  // ---- Stage 4: X1 = relu(X0 @ Wxfc1^T + b) -------------------------------
  {
    v8f a0 = {}, a1 = {};
    for (int kc = 0; kc < 8; ++kc) {
      v16bf a = lds_a_frag(s_X0, 256, kc * 32);
      WMMA_BF16(a0, a, load_b_tile(Bx1, kc * 16 + w));
      WMMA_BF16(a1, a, load_b_tile(Bx1, kc * 16 + w + 8));
    }
    store_c<true>(s_RB, 256, w * 16, a0, b_xfc1);
    store_c<true>(s_RB, 256, (w + 8) * 16, a1, b_xfc1);
  }
  __syncthreads();

  // ---- Stage 5: X = X1 @ Wxfc2^T + b (linear) -> back into s_X0 -----------
  {
    v8f a0 = {}, a1 = {};
    for (int kc = 0; kc < 8; ++kc) {
      v16bf a = lds_a_frag(s_RB, 256, kc * 32);
      WMMA_BF16(a0, a, load_b_tile(Bx2, kc * 16 + w));
      WMMA_BF16(a1, a, load_b_tile(Bx2, kc * 16 + w + 8));
    }
    store_c<false>(s_X0, 256, w * 16, a0, b_xfc2);
    store_c<false>(s_X0, 256, (w + 8) * 16, a1, b_xfc2);
  }
  __syncthreads();   // s_RB now dead -> reuse as per-wave h buffers

  unsigned short* hb = s_RB + w * 1024;   // 16x64 bf16 per wave
  unsigned short* lb = s_lb + w * 1024;   // 16x64 bf16 per wave

  // ---- Per-point phase: each wave owns points 2w and 2w+1 -----------------
  for (int pp = 0; pp < 2; ++pp) {
    const int pt  = w * 2 + pp;
    const int gpt = gp0 + pt;
    __builtin_prefetch(fts + (size_t)gpt * 16 * 64, 0, 0);  // stream fts tile

    // Stage 1: h = relu(ptsl_pt(16x3) @ Wfc1^T + b); one A, 4 indep tiles
    {
      v16bf a = make_a_frag([&](int m, int k) -> unsigned short {
        return k < 3 ? s_ptsl[pt * 64 + m * 3 + k] : (unsigned short)0;
      });
      v8f a0 = {}, a1 = {}, a2 = {}, a3 = {};
      WMMA_BF16(a0, a, load_b_tile(Bfc1, 0));
      WMMA_BF16(a1, a, load_b_tile(Bfc1, 1));
      WMMA_BF16(a2, a, load_b_tile(Bfc1, 2));
      WMMA_BF16(a3, a, load_b_tile(Bfc1, 3));
      store_c<true>(hb, 64, 0,  a0, b_fc1);
      store_c<true>(hb, 64, 16, a1, b_fc1);
      store_c<true>(hb, 64, 32, a2, b_fc1);
      store_c<true>(hb, 64, 48, a3, b_fc1);
    }

    // Stage 2: lifted = relu(h(16x64) @ Wfc2^T + b); pairs (nt, nt+2)
    for (int nt = 0; nt < 2; ++nt) {
      v8f a0 = {}, a1 = {};
      for (int kc = 0; kc < 2; ++kc) {
        v16bf a = lds_a_frag(hb, 64, kc * 32);
        WMMA_BF16(a0, a, load_b_tile(Bfc2, kc * 4 + nt));
        WMMA_BF16(a1, a, load_b_tile(Bfc2, kc * 4 + nt + 2));
      }
      store_c<true>(lb, 64, nt * 16, a0, b_fc2);
      store_c<true>(lb, 64, (nt + 2) * 16, a1, b_fc2);
    }

    // Stage 6: fts_X = X_pt(16x16) @ fts_cat(16x128)  (K padded 16->32)
    // Stage 7: depthwise reduce of fts_X straight out of the accumulators.
    {
      v16bf aX = lds_a_frag_k16(s_X0 + pt * 256, 16);   // loop-invariant A
      for (int cp = 0; cp < 4; ++cp) {
        v8f a0 = {}, a1 = {};
        v16bf b0 = make_b_frag([&](int k, int n) -> unsigned short {
          if (k >= 16) return (unsigned short)0;
          int c = cp * 16 + n;
          return c < 64 ? lb[k * 64 + c]
                        : f2bf(fts[(gpt * 16 + k) * 64 + (c - 64)]);
        });
        v16bf b1 = make_b_frag([&](int k, int n) -> unsigned short {
          if (k >= 16) return (unsigned short)0;
          int c = (cp + 4) * 16 + n;
          return c < 64 ? lb[k * 64 + c]
                        : f2bf(fts[(gpt * 16 + k) * 64 + (c - 64)]);
        });
        WMMA_BF16(a0, aX, b0);
        WMMA_BF16(a1, aX, b1);

#pragma unroll
        for (int half_sel = 0; half_sel < 2; ++half_sel) {
          const v8f& acc = half_sel ? a1 : a0;
          const int c = (cp + 4 * half_sel) * 16 + ln;   // channel this lane holds
#pragma unroll
          for (int m = 0; m < 4; ++m) {                  // depth multiplier
            const float4* wd = (const float4*)(W_dw + (c * 4 + m) * 16 + 8 * half);
            float4 w0 = wd[0], w1 = wd[1];
            float s = acc[0] * w0.x + acc[1] * w0.y + acc[2] * w0.z + acc[3] * w0.w +
                      acc[4] * w1.x + acc[5] * w1.y + acc[6] * w1.z + acc[7] * w1.w;
            s += __shfl_down(s, 16, 32);                 // combine lane-halves
            if (lane < 16)
              s_dw[pt * 512 + c * 4 + m] = f2bf(s + b_dw[c * 4 + m]);
          }
        }
      }
    }
  }
  __syncthreads();

  // ---- Stage 8: y = dw(16x512) @ Wpw^T(512x128) + b; wave w -> 16 channels
  {
    const int c0 = w << 4;
    v8f a0 = {}, a1 = {};                         // split K for independence
    for (int kc = 0; kc < 16; kc += 2) {
      v16bf x0 = lds_a_frag(s_dw, 512, kc * 32);
      v16bf x1 = lds_a_frag(s_dw, 512, (kc + 1) * 32);
      WMMA_BF16(a0, x0, load_b_tile(Bpw, kc * 8 + w));
      WMMA_BF16(a1, x1, load_b_tile(Bpw, (kc + 1) * 8 + w));
    }
    const int c = c0 + ln;
    const float bb = b_pw[c];
    float s1 = 0.0f, s2 = 0.0f;
#pragma unroll
    for (int v = 0; v < 8; ++v) {
      int pt = v + 8 * half;
      float val = a0[v] + a1[v] + bb;
      out[(gp0 + pt) * 128 + c] = val;   // pre-BN, already (N,P,Cout) layout
      s1 += val;
      s2 += val * val;
    }
    s1 += __shfl_down(s1, 16, 32);
    s2 += __shfl_down(s2, 16, 32);
    if (lane < 16) {
      partials[(tile * 128 + c) * 2 + 0] = s1;
      partials[(tile * 128 + c) * 2 + 1] = s2;
    }
  }
}

// ---------------------------------------------------------------------------
// BN: deterministic sequential reduction over the 1024 per-tile partials.
// ---------------------------------------------------------------------------
__global__ void bn_finalize(const float* __restrict__ partials,
                            const float* __restrict__ gamma,
                            const float* __restrict__ beta,
                            float* __restrict__ stats) {
  int ch = threadIdx.x;                 // 128 threads
  float s1 = 0.0f, s2 = 0.0f;
  for (int t = 0; t < 1024; ++t) {
    s1 += partials[(t * 128 + ch) * 2 + 0];
    s2 += partials[(t * 128 + ch) * 2 + 1];
  }
  const float inv = 1.0f / 16384.0f;
  float mean  = s1 * inv;
  float var   = s2 * inv - mean * mean;
  float scale = gamma[ch] * rsqrtf(var + 1e-5f);
  stats[ch]        = scale;
  stats[128 + ch]  = beta[ch] - mean * scale;
}

__global__ void bn_apply(float* __restrict__ out,
                         const float* __restrict__ stats) {
  int idx = blockIdx.x * 256 + threadIdx.x;   // over float4s: 524288 total
  float4* o4 = (float4*)out;
  float4 v = o4[idx];
  int c0 = (idx * 4) & 127;
  v.x = fmaxf(v.x * stats[c0 + 0] + stats[128 + c0 + 0], 0.0f);
  v.y = fmaxf(v.y * stats[c0 + 1] + stats[128 + c0 + 1], 0.0f);
  v.z = fmaxf(v.z * stats[c0 + 2] + stats[128 + c0 + 2], 0.0f);
  v.w = fmaxf(v.w * stats[c0 + 3] + stats[128 + c0 + 3], 0.0f);
  o4[idx] = v;
}

// ---------------------------------------------------------------------------
extern "C" void kernel_launch(void* const* d_in, const int* in_sizes, int n_in,
                              void* d_out, int out_size, void* d_ws, size_t ws_size,
                              hipStream_t stream) {
  const float* rep_pt  = (const float*)d_in[0];
  const float* pts     = (const float*)d_in[1];
  const float* fts     = (const float*)d_in[2];
  const float* W_fc1   = (const float*)d_in[3];
  const float* b_fc1   = (const float*)d_in[4];
  const float* W_fc2   = (const float*)d_in[5];
  const float* b_fc2   = (const float*)d_in[6];
  const float* W_xconv = (const float*)d_in[7];
  const float* b_xconv = (const float*)d_in[8];
  const float* W_xfc1  = (const float*)d_in[9];
  const float* b_xfc1  = (const float*)d_in[10];
  const float* W_xfc2  = (const float*)d_in[11];
  const float* b_xfc2  = (const float*)d_in[12];
  const float* W_dw    = (const float*)d_in[13];
  const float* b_dw    = (const float*)d_in[14];
  const float* W_pw    = (const float*)d_in[15];
  const float* b_pw    = (const float*)d_in[16];
  const float* gamma   = (const float*)d_in[17];
  const float* beta    = (const float*)d_in[18];

  unsigned short* wsw = (unsigned short*)d_ws;
  float* stats    = (float*)((char*)d_ws + OFF_STAT_BYTES);
  float* partials = (float*)((char*)d_ws + OFF_PART_BYTES);
  float* out      = (float*)d_out;

  prep_weights<<<(N_WCONV_ELEMS + 255) / 256, 256, 0, stream>>>(
      W_fc1, W_fc2, W_xconv, W_xfc1, W_xfc2, W_pw, wsw);

  xconv_main<<<1024, 256, 0, stream>>>(
      rep_pt, pts, fts, b_fc1, b_fc2, b_xconv, b_xfc1, b_xfc2,
      W_dw, b_dw, b_pw, wsw, partials, out);

  bn_finalize<<<1, 128, 0, stream>>>(partials, gamma, beta, stats);

  bn_apply<<<(16 * 1024 * 128) / 4 / 256, 256, 0, stream>>>(out, stats);
}
